// CrossPVT_T2T_MambaDINO_87651692577382
// MI455X (gfx1250) — matrix operand order repaired and checked by reading, not compiled, not run on hardware
//
#include <hip/hip_runtime.h>

// ---------------------------------------------------------------------------
// Constants (reference shapes)
// ---------------------------------------------------------------------------
constexpr int DIMc  = 384;
constexpr int HEADSc= 6;
constexpr int HDc   = 64;
constexpr int Bc    = 8;
constexpr int NSEQ  = 1025;   // 1 CLS + 1024 tokens
constexpr int NPADc = 1056;   // padded to 33 * 32
constexpr int Mrows = Bc * NPADc;       // 8448 rows (divisible by 256)
constexpr int HIDc  = 1152;
constexpr int FC1c  = 2304;

// ---------------------------------------------------------------------------
// WMMA fragment types (gfx1250, wave32)
// ---------------------------------------------------------------------------
typedef __bf16        bf16x16 __attribute__((ext_vector_type(16)));
typedef float         f32x8   __attribute__((ext_vector_type(8)));
typedef unsigned int  u32x4   __attribute__((ext_vector_type(4)));

union FragB { bf16x16 v; u32x4 q[2]; unsigned short u[16]; };
union FragF { f32x8   v; float f[8]; };

__device__ __forceinline__ unsigned short f2bfbits(float f) {
    union { float f; unsigned u; } x; x.f = f;
    unsigned r = x.u + 0x7FFFu + ((x.u >> 16) & 1u);   // RNE
    return (unsigned short)(r >> 16);
}

// ---------------------------------------------------------------------------
// fp32 -> bf16 elementwise
// ---------------------------------------------------------------------------
__global__ __launch_bounds__(256) void k_f32_to_bf16(
    const float* __restrict__ src, unsigned short* __restrict__ dst, long long n)
{
    long long i = (long long)blockIdx.x * 256 + threadIdx.x;
    if (i < n) dst[i] = f2bfbits(src[i]);
}

// ---------------------------------------------------------------------------
// fp32 [K,N] -> bf16 transposed [N,K]  (one-time weight prep; coalesced reads)
// ---------------------------------------------------------------------------
__global__ __launch_bounds__(256) void k_f32_to_bf16_T(
    const float* __restrict__ src, unsigned short* __restrict__ dst,
    int K, int N)
{
    long long i = (long long)blockIdx.x * 256 + threadIdx.x;
    if (i >= (long long)K * N) return;
    int k = (int)(i / N), n = (int)(i % N);
    dst[(size_t)n * K + k] = f2bfbits(src[i]);
}

// ---------------------------------------------------------------------------
// Pack: [B,1024,384] -> padded [B,NPAD,384] fp32 + bf16 (CLS row 0 = zeros)
// ---------------------------------------------------------------------------
__global__ __launch_bounds__(256) void k_pack(
    const float* __restrict__ src, float* __restrict__ dst,
    unsigned short* __restrict__ dst_bf)
{
    long long i = (long long)blockIdx.x * 256 + threadIdx.x;   // over Mrows*384
    if (i >= (long long)Mrows * DIMc) return;
    int row = (int)(i / DIMc), c = (int)(i % DIMc);
    int b = row / NPADc, n = row % NPADc;
    float v = 0.f;
    if (n >= 1 && n <= 1024)
        v = src[((size_t)b * 1024 + (n - 1)) * DIMc + c];
    dst[i] = v;
    dst_bf[i] = f2bfbits(v);
}

// ---------------------------------------------------------------------------
// Unpack: concat(tokS[:, :1025], tokB[:, :1025]) -> out [B,2050,384] fp32
// ---------------------------------------------------------------------------
__global__ __launch_bounds__(256) void k_unpack(
    const float* __restrict__ tokS, const float* __restrict__ tokB,
    float* __restrict__ out)
{
    long long i = (long long)blockIdx.x * 256 + threadIdx.x;   // B*2050*384
    if (i >= (long long)Bc * 2 * NSEQ * DIMc) return;
    int c = (int)(i % DIMc);
    long long row = i / DIMc;
    int b = (int)(row / (2 * NSEQ));
    int n = (int)(row % (2 * NSEQ));
    const float* srcp = (n < NSEQ)
        ? tokS + ((size_t)b * NPADc + n) * DIMc + c
        : tokB + ((size_t)b * NPADc + (n - NSEQ)) * DIMc + c;
    out[i] = *srcp;
}

// ---------------------------------------------------------------------------
// LayerNorm over DIM=384; one wave per row; bf16 output (GEMM operand)
// ---------------------------------------------------------------------------
__global__ __launch_bounds__(256) void k_layernorm(
    const float* __restrict__ x, const float* __restrict__ g,
    const float* __restrict__ bta, unsigned short* __restrict__ out, int rows)
{
    int row = blockIdx.x * 8 + (threadIdx.x >> 5);
    int lane = threadIdx.x & 31;
    if (row >= rows) return;
    const float* xr = x + (size_t)row * DIMc;
    float v[12]; float s = 0.f;
    #pragma unroll
    for (int i = 0; i < 12; ++i) { v[i] = xr[lane + i * 32]; s += v[i]; }
    #pragma unroll
    for (int off = 16; off; off >>= 1) s += __shfl_xor(s, off, 32);
    float mean = s * (1.f / DIMc);
    float q = 0.f;
    #pragma unroll
    for (int i = 0; i < 12; ++i) { float d = v[i] - mean; q += d * d; }
    #pragma unroll
    for (int off = 16; off; off >>= 1) q += __shfl_xor(q, off, 32);
    float inv = rsqrtf(q * (1.f / DIMc) + 1e-5f);
    unsigned short* orow = out + (size_t)row * DIMc;
    #pragma unroll
    for (int i = 0; i < 12; ++i) {
        int c = lane + i * 32;
        orow[c] = f2bfbits((v[i] - mean) * inv * g[c] + bta[c]);
    }
}

// ---------------------------------------------------------------------------
// LDS-free, software-pipelined WMMA GEMM:
//   C[M,N] = A_bf16[M,K] @ Wt_bf16[N,K]^T + bias
// Block = 256 threads (8 waves); block tile 256x64; wave tile 32x64.
// Double-buffered fragment sets so loads for step k+1 stay in flight while
// the 8 WMMAs of step k issue (partial s_wait_loadcnt instead of full drain).
// ---------------------------------------------------------------------------
__global__ __launch_bounds__(256) void k_gemm_bf16(
    const unsigned short* __restrict__ A,   // [M][K]
    const unsigned short* __restrict__ Wt,  // [N][K]  (transposed weights)
    const float* __restrict__ bias, float* __restrict__ C, int K, int N)
{
    const int wave = threadIdx.x >> 5;
    const int lane = threadIdx.x & 31;
    const int bm = blockIdx.y * 256 + wave * 32;
    const int bn = blockIdx.x * 64;

    FragF o[2][4];
    #pragma unroll
    for (int nt = 0; nt < 4; ++nt) {
        float bv = bias[bn + nt * 16 + (lane & 15)];
        #pragma unroll
        for (int r = 0; r < 8; ++r) { o[0][nt].f[r] = bv; o[1][nt].f[r] = bv; }
    }

    const int akb = (lane < 16) ? 0 : 8;     // A fragment K-base
    const int bkb = (lane < 16) ? 0 : 8;     // B fragment K-base
    const unsigned short* arow0 = A + (size_t)(bm + (lane & 15)) * K + akb;
    const unsigned short* arow1 = arow0 + (size_t)16 * K;
    const unsigned short* brow[4];
    #pragma unroll
    for (int nt = 0; nt < 4; ++nt)
        brow[nt] = Wt + (size_t)(bn + nt * 16 + (lane & 15)) * K + bkb;

    struct Frags { FragB a0, a1, b[4]; };
    auto load_set = [&](int kt, Frags& f) {
        f.a0.q[0] = *(const u32x4*)(arow0 + kt);
        f.a0.q[1] = *(const u32x4*)(arow0 + kt + 16);
        f.a1.q[0] = *(const u32x4*)(arow1 + kt);
        f.a1.q[1] = *(const u32x4*)(arow1 + kt + 16);
        #pragma unroll
        for (int nt = 0; nt < 4; ++nt) {
            f.b[nt].q[0] = *(const u32x4*)(brow[nt] + kt);
            f.b[nt].q[1] = *(const u32x4*)(brow[nt] + kt + 16);
        }
    };
    auto compute = [&](Frags& f) {
        #pragma unroll
        for (int nt = 0; nt < 4; ++nt) {
            o[0][nt].v = __builtin_amdgcn_wmma_f32_16x16x32_bf16(
                false, f.a0.v, false, f.b[nt].v, (short)0, o[0][nt].v, false, false);
            o[1][nt].v = __builtin_amdgcn_wmma_f32_16x16x32_bf16(
                false, f.a1.v, false, f.b[nt].v, (short)0, o[1][nt].v, false, false);
        }
    };

    Frags f0, f1;
    load_set(0, f0);
    for (int kt = 0; kt < K; kt += 64) {      // K is a multiple of 64 here
        load_set(kt + 32, f1);
        compute(f0);
        if (kt + 64 < K) load_set(kt + 64, f0);
        compute(f1);
    }

    #pragma unroll
    for (int mt = 0; mt < 2; ++mt) {
        #pragma unroll
        for (int nt = 0; nt < 4; ++nt) {
            int c = bn + nt * 16 + (lane & 15);
            #pragma unroll
            for (int r = 0; r < 8; ++r) {
                int m = bm + mt * 16 + ((lane < 16) ? r : r + 8);
                C[(size_t)m * N + c] = o[mt][nt].f[r];
            }
        }
    }
}

// ---------------------------------------------------------------------------
// Repack Q: [M,384] fp32 -> [B*H][NPAD][64] bf16, optional per-(row,head) l2norm
// ---------------------------------------------------------------------------
__global__ __launch_bounds__(256) void k_repack_q(
    const float* __restrict__ q, unsigned short* __restrict__ qh, int do_l2)
{
    int idx = blockIdx.x * 8 + (threadIdx.x >> 5);
    if (idx >= Mrows * HEADSc) return;
    int lane = threadIdx.x & 31;
    int r = idx / HEADSc, h = idx % HEADSc;
    const float* src = q + (size_t)r * DIMc + h * HDc;
    float a0 = src[lane], a1 = src[lane + 32];
    if (do_l2) {
        float ss = a0 * a0 + a1 * a1;
        #pragma unroll
        for (int off = 16; off; off >>= 1) ss += __shfl_xor(ss, off, 32);
        float inv = 1.f / fmaxf(sqrtf(ss), 1e-12f);
        a0 *= inv; a1 *= inv;
    }
    int b = r / NPADc, n = r % NPADc;
    size_t dst = ((size_t)(b * HEADSc + h) * NPADc + n) * HDc;
    qh[dst + lane]      = f2bfbits(a0);
    qh[dst + lane + 32] = f2bfbits(a1);
}

// ---------------------------------------------------------------------------
// Repack KV: [M,768] fp32 -> K [B*H][NPAD][64] bf16 (opt l2norm)
//                            Vt [B*H][64][NPAD] bf16 (transposed)
// ---------------------------------------------------------------------------
__global__ __launch_bounds__(256) void k_repack_kv(
    const float* __restrict__ kv, unsigned short* __restrict__ kh,
    unsigned short* __restrict__ vth, int do_l2)
{
    int idx = blockIdx.x * 8 + (threadIdx.x >> 5);
    if (idx >= Mrows * HEADSc) return;
    int lane = threadIdx.x & 31;
    int r = idx / HEADSc, h = idx % HEADSc;
    const float* row = kv + (size_t)r * (2 * DIMc);
    float k0 = row[h * HDc + lane],        k1 = row[h * HDc + lane + 32];
    float v0 = row[DIMc + h * HDc + lane], v1 = row[DIMc + h * HDc + lane + 32];
    if (do_l2) {
        float ss = k0 * k0 + k1 * k1;
        #pragma unroll
        for (int off = 16; off; off >>= 1) ss += __shfl_xor(ss, off, 32);
        float inv = 1.f / fmaxf(sqrtf(ss), 1e-12f);
        k0 *= inv; k1 *= inv;
    }
    int b = r / NPADc, n = r % NPADc;
    size_t kb = ((size_t)(b * HEADSc + h) * NPADc + n) * HDc;
    kh[kb + lane]      = f2bfbits(k0);
    kh[kb + lane + 32] = f2bfbits(k1);
    size_t vb = (size_t)(b * HEADSc + h) * HDc * NPADc;
    vth[vb + (size_t)lane * NPADc + n]        = f2bfbits(v0);
    vth[vb + (size_t)(lane + 32) * NPADc + n] = f2bfbits(v1);
}

// ---------------------------------------------------------------------------
// Flash attention: one wave per (b,h, 16-query tile).
// All K and V fragments for a key tile are loaded up front; the softmax VALU
// chain then hides the V-load latency before the P*V WMMAs. Keys>=1025 masked.
// ---------------------------------------------------------------------------
__global__ __launch_bounds__(256) void k_attn(
    const unsigned short* __restrict__ qh, const unsigned short* __restrict__ kh,
    const unsigned short* __restrict__ vth, unsigned short* __restrict__ attn_out,
    const float* __restrict__ scale_ptr, float scale_const)
{
    __shared__ __align__(16) unsigned short lP[8][16][32];
    const int QT = NPADc / 16;                      // 66
    int wave = threadIdx.x >> 5, lane = threadIdx.x & 31;
    int task = blockIdx.x * 8 + wave;
    if (task >= Bc * HEADSc * QT) return;
    int bh = task / QT, qt = task % QT;
    int h = bh % HEADSc, b = bh / HEADSc;
    float scale = scale_ptr ? scale_ptr[h] : scale_const;
    int q0 = qt * 16;

    const unsigned short* qbase = qh  + (size_t)bh * NPADc * HDc;
    const unsigned short* kbase = kh  + (size_t)bh * NPADc * HDc;
    const unsigned short* vbase = vth + (size_t)bh * HDc * NPADc;

    // Q fragments for K-steps 0 (dims 0..31) and 1 (dims 32..63)
    FragB qf[2];
    {
        int m = lane & 15;
        const unsigned short* qr = qbase + (size_t)(q0 + m) * HDc;
        int kb = (lane < 16) ? 0 : 8;
        qf[0].q[0] = *(const u32x4*)(qr + kb);
        qf[0].q[1] = *(const u32x4*)(qr + kb + 16);
        qf[1].q[0] = *(const u32x4*)(qr + 32 + kb);
        qf[1].q[1] = *(const u32x4*)(qr + 32 + kb + 16);
    }

    FragF o[4];
    #pragma unroll
    for (int nt = 0; nt < 4; ++nt)
        #pragma unroll
        for (int r = 0; r < 8; ++r) o[nt].f[r] = 0.f;
    float mrow[8], lrow[8];
    #pragma unroll
    for (int r = 0; r < 8; ++r) { mrow[r] = -1e30f; lrow[r] = 0.f; }

    for (int k0 = 0; k0 < NPADc; k0 += 32) {
        // ---- hoist ALL fragment loads for this key tile ----
        FragB kf[2][2];                 // [col-tile][k-step]
        #pragma unroll
        for (int ct = 0; ct < 2; ++ct) {
            const unsigned short* kr =
                kbase + (size_t)(k0 + ct * 16 + (lane & 15)) * HDc;
            int kb2 = (lane < 16) ? 0 : 8;
            #pragma unroll
            for (int ks = 0; ks < 2; ++ks) {
                kf[ct][ks].q[0] = *(const u32x4*)(kr + ks * 32 + kb2);
                kf[ct][ks].q[1] = *(const u32x4*)(kr + ks * 32 + kb2 + 16);
            }
        }
        FragB vf[4];                    // V fragments; consumed only at the end
        #pragma unroll
        for (int nt = 0; nt < 4; ++nt) {
            const unsigned short* vr = vbase
                + (size_t)(nt * 16 + (lane & 15)) * NPADc + k0 + ((lane < 16) ? 0 : 8);
            vf[nt].q[0] = *(const u32x4*)(vr);
            vf[nt].q[1] = *(const u32x4*)(vr + 16);
        }

        // ---- S = Q K^T (16x32 tile) ----
        FragF s0, s1;
        #pragma unroll
        for (int r = 0; r < 8; ++r) { s0.f[r] = 0.f; s1.f[r] = 0.f; }
        #pragma unroll
        for (int ks = 0; ks < 2; ++ks) {
            s0.v = __builtin_amdgcn_wmma_f32_16x16x32_bf16(
                false, qf[ks].v, false, kf[0][ks].v, (short)0, s0.v, false, false);
            s1.v = __builtin_amdgcn_wmma_f32_16x16x32_bf16(
                false, qf[ks].v, false, kf[1][ks].v, (short)0, s1.v, false, false);
        }

        // ---- scale + key mask + online softmax update ----
        int key0 = k0 + (lane & 15);
        int key1 = k0 + 16 + (lane & 15);
        bool ok0 = key0 < NSEQ, ok1 = key1 < NSEQ;
        float alpha[8];
        #pragma unroll
        for (int r = 0; r < 8; ++r) {
            s0.f[r] = ok0 ? s0.f[r] * scale : -1e30f;
            s1.f[r] = ok1 ? s1.f[r] * scale : -1e30f;
            float t = fmaxf(s0.f[r], s1.f[r]);
            t = fmaxf(t, __shfl_xor(t, 1, 32));
            t = fmaxf(t, __shfl_xor(t, 2, 32));
            t = fmaxf(t, __shfl_xor(t, 4, 32));
            t = fmaxf(t, __shfl_xor(t, 8, 32));
            float mnew = fmaxf(mrow[r], t);
            alpha[r] = __expf(mrow[r] - mnew);
            mrow[r] = mnew;
            s0.f[r] = __expf(s0.f[r] - mnew);
            s1.f[r] = __expf(s1.f[r] - mnew);
            float rs = s0.f[r] + s1.f[r];
            rs += __shfl_xor(rs, 1, 32);
            rs += __shfl_xor(rs, 2, 32);
            rs += __shfl_xor(rs, 4, 32);
            rs += __shfl_xor(rs, 8, 32);
            lrow[r] = lrow[r] * alpha[r] + rs;
        }
        #pragma unroll
        for (int nt = 0; nt < 4; ++nt)
            #pragma unroll
            for (int r = 0; r < 8; ++r) o[nt].f[r] *= alpha[r];

        // ---- relayout P (C-layout) -> A-fragment via wave-private LDS ----
        {
            int nn = lane & 15;
            int rb = (lane < 16) ? 0 : 8;
            #pragma unroll
            for (int r = 0; r < 8; ++r) {
                lP[wave][rb + r][nn]      = f2bfbits(s0.f[r]);
                lP[wave][rb + r][16 + nn] = f2bfbits(s1.f[r]);
            }
        }
        FragB pf;
        {
            int m = lane & 15;
            int kb = (lane < 16) ? 0 : 8;
            pf.q[0] = *(const u32x4*)&lP[wave][m][kb];
            pf.q[1] = *(const u32x4*)&lP[wave][m][kb + 16];
        }
        // ---- O += P V (V fragments already resident) ----
        #pragma unroll
        for (int nt = 0; nt < 4; ++nt) {
            o[nt].v = __builtin_amdgcn_wmma_f32_16x16x32_bf16(
                false, pf.v, false, vf[nt].v, (short)0, o[nt].v, false, false);
        }
    }

    // epilogue: normalize and merge heads -> [B][NPAD][384] bf16
    unsigned short* obase =
        attn_out + ((size_t)b * NPADc + q0) * DIMc + h * HDc;
    #pragma unroll
    for (int nt = 0; nt < 4; ++nt) {
        int c = nt * 16 + (lane & 15);
        #pragma unroll
        for (int r = 0; r < 8; ++r) {
            int m = (lane < 16) ? r : r + 8;
            obase[(size_t)m * DIMc + c] = f2bfbits(o[nt].f[r] / lrow[r]);
        }
    }
}

// ---------------------------------------------------------------------------
// GEGLU: h = x1 * gelu_exact(gate);  fc1out [M,2304] -> bf16 [M,1152]
// ---------------------------------------------------------------------------
__global__ __launch_bounds__(256) void k_geglu(
    const float* __restrict__ h, unsigned short* __restrict__ out, long long n)
{
    long long i = (long long)blockIdx.x * 256 + threadIdx.x;
    if (i >= n) return;
    long long row = i / HIDc; int c = (int)(i % HIDc);
    float x1 = h[row * FC1c + c];
    float g  = h[row * FC1c + HIDc + c];
    float ge = 0.5f * g * (1.f + erff(g * 0.70710678118654752f));
    out[i] = f2bfbits(x1 * ge);
}

// ---------------------------------------------------------------------------
// Residual with layer-scale: tok += ls[c] * x ; optional bf16 shadow
// ---------------------------------------------------------------------------
__global__ __launch_bounds__(256) void k_residual(
    float* __restrict__ tok, const float* __restrict__ x,
    const float* __restrict__ lsrow, unsigned short* __restrict__ tok_bf,
    long long n)
{
    long long i = (long long)blockIdx.x * 256 + threadIdx.x;
    if (i >= n) return;
    int c = (int)(i % DIMc);
    float v = tok[i] + lsrow[c] * x[i];
    tok[i] = v;
    if (tok_bf) tok_bf[i] = f2bfbits(v);
}

// ===========================================================================
// Host orchestration
// ===========================================================================
extern "C" void kernel_launch(void* const* d_in, const int* in_sizes, int n_in,
                              void* d_out, int out_size, void* d_ws, size_t ws_size,
                              hipStream_t stream)
{
    (void)in_sizes; (void)n_in; (void)out_size; (void)ws_size;
    char* base = (char*)d_ws;
    size_t off = 0;
    auto alloc = [&](size_t bytes) -> void* {
        void* p = base + off;
        off += (bytes + 255) & ~(size_t)255;
        return p;
    };

    const size_t MD = (size_t)Mrows * DIMc;
    float*          tokS    = (float*)alloc(MD * 4);
    float*          tokB    = (float*)alloc(MD * 4);
    unsigned short* tokSbf  = (unsigned short*)alloc(MD * 2);
    unsigned short* tokBbf  = (unsigned short*)alloc(MD * 2);
    unsigned short* normbf  = (unsigned short*)alloc(MD * 2);
    float*          qf32    = (float*)alloc(MD * 4);
    float*          kvf32   = (float*)alloc(MD * 2 * 4);
    unsigned short* qh      = (unsigned short*)alloc(MD * 2);
    unsigned short* kh      = (unsigned short*)alloc(MD * 2);
    unsigned short* vth     = (unsigned short*)alloc(MD * 2);
    unsigned short* attnbf  = (unsigned short*)alloc(MD * 2);
    float*          gemmC   = (float*)alloc((size_t)Mrows * FC1c * 4);
    unsigned short* geglubf = (unsigned short*)alloc((size_t)Mrows * HIDc * 2);
    float*          crossA  = (float*)alloc(MD * 4);
    float*          crossB  = (float*)alloc(MD * 4);

    // one-time: fp32 [K,N] weights -> bf16 transposed [N,K]
    auto convT = [&](const float* w, int K, int N) -> unsigned short* {
        unsigned short* dst = (unsigned short*)alloc((size_t)K * N * 2);
        long long tot = (long long)K * N;
        k_f32_to_bf16_T<<<(unsigned)((tot + 255) / 256), 256, 0, stream>>>(
            w, dst, K, N);
        return dst;
    };

    // ---- parse params (jax pytree: dict keys sorted alphabetically) ----
    struct LayerW {
        const float *ca_kva_b, *ca_kvb_b, *ca_pa_b, *ca_pb_b, *ca_qa_b, *ca_qb_b;
        unsigned short *ca_kva_w, *ca_kvb_w, *ca_pa_w, *ca_pb_w, *ca_qa_w, *ca_qb_w;
        const float *ffnb_fc1_b, *ffnb_fc2_b, *ffns_fc1_b, *ffns_fc2_b;
        unsigned short *ffnb_fc1_w, *ffnb_fc2_w, *ffns_fc1_w, *ffns_fc2_w;
        const float *ls;
        const float *nb_b, *nb_g, *nb2_b, *nb2_g, *ns_b, *ns_g, *ns2_b, *ns2_g;
        const float *sab_kv_b, *sab_p_b, *sab_q_b, *sab_scale;
        unsigned short *sab_kv_w, *sab_p_w, *sab_q_w;
        const float *sas_kv_b, *sas_p_b, *sas_q_b, *sas_scale;
        unsigned short *sas_kv_w, *sas_p_w, *sas_q_w;
    } L[2];

    for (int l = 0; l < 2; ++l) {
        void* const* p = d_in + 2 + l * 43;
        auto Fp = [&](int i) { return (const float*)p[i]; };
        L[l].ca_kva_b = Fp(0);  L[l].ca_kva_w = convT(Fp(1),  DIMc, 2 * DIMc);
        L[l].ca_kvb_b = Fp(2);  L[l].ca_kvb_w = convT(Fp(3),  DIMc, 2 * DIMc);
        L[l].ca_pa_b  = Fp(4);  L[l].ca_pa_w  = convT(Fp(5),  DIMc, DIMc);
        L[l].ca_pb_b  = Fp(6);  L[l].ca_pb_w  = convT(Fp(7),  DIMc, DIMc);
        L[l].ca_qa_b  = Fp(8);  L[l].ca_qa_w  = convT(Fp(9),  DIMc, DIMc);
        L[l].ca_qb_b  = Fp(10); L[l].ca_qb_w  = convT(Fp(11), DIMc, DIMc);
        L[l].ffnb_fc1_b = Fp(12); L[l].ffnb_fc1_w = convT(Fp(13), DIMc, FC1c);
        L[l].ffnb_fc2_b = Fp(14); L[l].ffnb_fc2_w = convT(Fp(15), HIDc, DIMc);
        L[l].ffns_fc1_b = Fp(16); L[l].ffns_fc1_w = convT(Fp(17), DIMc, FC1c);
        L[l].ffns_fc2_b = Fp(18); L[l].ffns_fc2_w = convT(Fp(19), HIDc, DIMc);
        L[l].ls = Fp(20);
        L[l].nb_b  = Fp(21); L[l].nb_g  = Fp(22);
        L[l].nb2_b = Fp(23); L[l].nb2_g = Fp(24);
        L[l].ns_b  = Fp(25); L[l].ns_g  = Fp(26);
        L[l].ns2_b = Fp(27); L[l].ns2_g = Fp(28);
        L[l].sab_kv_b = Fp(29); L[l].sab_kv_w = convT(Fp(30), DIMc, 2 * DIMc);
        L[l].sab_p_b  = Fp(31); L[l].sab_p_w  = convT(Fp(32), DIMc, DIMc);
        L[l].sab_q_b  = Fp(33); L[l].sab_q_w  = convT(Fp(34), DIMc, DIMc);
        L[l].sab_scale = Fp(35);
        L[l].sas_kv_b = Fp(36); L[l].sas_kv_w = convT(Fp(37), DIMc, 2 * DIMc);
        L[l].sas_p_b  = Fp(38); L[l].sas_p_w  = convT(Fp(39), DIMc, DIMc);
        L[l].sas_q_b  = Fp(40); L[l].sas_q_w  = convT(Fp(41), DIMc, DIMc);
        L[l].sas_scale = Fp(42);
    }

    // ---- launch helpers ----
    const unsigned EW = (unsigned)((MD + 255) / 256);                 // 12672
    const unsigned LN = (unsigned)(Mrows / 8);                        // 1056
    const unsigned RP = (unsigned)((Mrows * HEADSc) / 8);             // 6336
    const unsigned AT = (unsigned)((Bc * HEADSc * (NPADc / 16)) / 8); // 396
    const unsigned GG = (unsigned)(((size_t)Mrows * HIDc + 255) / 256);

    auto gemm = [&](const unsigned short* A, const unsigned short* Wt,
                    const float* bias, float* C, int K, int N) {
        dim3 grid((unsigned)(N / 64), (unsigned)(Mrows / 256));       // 33 rows of blocks
        k_gemm_bf16<<<grid, 256, 0, stream>>>(A, Wt, bias, C, K, N);
    };
    auto run_attn = [&](const unsigned short* qin, const unsigned short* kvin,
                        unsigned short* Wq, const float* bq,
                        unsigned short* Wkv, const float* bkv,
                        int do_l2, const float* scale_ptr) {
        gemm(qin,  Wq,  bq,  qf32,  DIMc, DIMc);
        gemm(kvin, Wkv, bkv, kvf32, DIMc, 2 * DIMc);
        k_repack_q <<<RP, 256, 0, stream>>>(qf32, qh, do_l2);
        k_repack_kv<<<RP, 256, 0, stream>>>(kvf32, kh, vth, do_l2);
        k_attn<<<AT, 256, 0, stream>>>(qh, kh, vth, attnbf, scale_ptr, 0.125f);
    };

    // ---- pack inputs ----
    k_pack<<<EW, 256, 0, stream>>>((const float*)d_in[0], tokS, tokSbf);
    k_pack<<<EW, 256, 0, stream>>>((const float*)d_in[1], tokB, tokBbf);

    for (int l = 0; l < 2; ++l) {
        LayerW& w = L[l];
        const float* ls0 = w.ls + 0 * DIMc;
        const float* ls1 = w.ls + 1 * DIMc;
        const float* ls2 = w.ls + 2 * DIMc;

        // --- cosine self-attention, stream S ---
        k_layernorm<<<LN, 256, 0, stream>>>(tokS, w.ns_g, w.ns_b, normbf, Mrows);
        run_attn(normbf, normbf, w.sas_q_w, w.sas_q_b, w.sas_kv_w, w.sas_kv_b,
                 1, w.sas_scale);
        gemm(attnbf, w.sas_p_w, w.sas_p_b, crossA, DIMc, DIMc);
        k_residual<<<EW, 256, 0, stream>>>(tokS, crossA, ls0, tokSbf, (long long)MD);

        // --- cosine self-attention, stream B ---
        k_layernorm<<<LN, 256, 0, stream>>>(tokB, w.nb_g, w.nb_b, normbf, Mrows);
        run_attn(normbf, normbf, w.sab_q_w, w.sab_q_b, w.sab_kv_w, w.sab_kv_b,
                 1, w.sab_scale);
        gemm(attnbf, w.sab_p_w, w.sab_p_b, crossB, DIMc, DIMc);
        k_residual<<<EW, 256, 0, stream>>>(tokB, crossB, ls0, tokBbf, (long long)MD);

        // --- cross attention (both directions computed before residuals) ---
        run_attn(tokSbf, tokBbf, w.ca_qa_w, w.ca_qa_b, w.ca_kvb_w, w.ca_kvb_b,
                 0, nullptr);
        gemm(attnbf, w.ca_pa_w, w.ca_pa_b, crossA, DIMc, DIMc);
        run_attn(tokBbf, tokSbf, w.ca_qb_w, w.ca_qb_b, w.ca_kva_w, w.ca_kva_b,
                 0, nullptr);
        gemm(attnbf, w.ca_pb_w, w.ca_pb_b, crossB, DIMc, DIMc);
        k_residual<<<EW, 256, 0, stream>>>(tokS, crossA, ls1, tokSbf, (long long)MD);
        k_residual<<<EW, 256, 0, stream>>>(tokB, crossB, ls1, tokBbf, (long long)MD);

        // --- FFN (GEGLU), stream S ---
        k_layernorm<<<LN, 256, 0, stream>>>(tokS, w.ns2_g, w.ns2_b, normbf, Mrows);
        gemm(normbf, w.ffns_fc1_w, w.ffns_fc1_b, gemmC, DIMc, FC1c);
        k_geglu<<<GG, 256, 0, stream>>>(gemmC, geglubf, (long long)Mrows * HIDc);
        gemm(geglubf, w.ffns_fc2_w, w.ffns_fc2_b, crossA, HIDc, DIMc);
        k_residual<<<EW, 256, 0, stream>>>(tokS, crossA, ls2, tokSbf, (long long)MD);

        // --- FFN (GEGLU), stream B ---
        k_layernorm<<<LN, 256, 0, stream>>>(tokB, w.nb2_g, w.nb2_b, normbf, Mrows);
        gemm(normbf, w.ffnb_fc1_w, w.ffnb_fc1_b, gemmC, DIMc, FC1c);
        k_geglu<<<GG, 256, 0, stream>>>(gemmC, geglubf, (long long)Mrows * HIDc);
        gemm(geglubf, w.ffnb_fc2_w, w.ffnb_fc2_b, crossB, HIDc, DIMc);
        k_residual<<<EW, 256, 0, stream>>>(tokB, crossB, ls2, tokBbf, (long long)MD);
    }

    // ---- unpack to [B, 2050, 384] fp32 ----
    const unsigned UP = (unsigned)(((size_t)Bc * 2 * NSEQ * DIMc + 255) / 256);
    k_unpack<<<UP, 256, 0, stream>>>(tokS, tokB, (float*)d_out);
}
